// e3d_lstm_cell_58428735095521
// MI455X (gfx1250) — matrix operand order repaired and checked by reading, not compiled, not run on hardware
//
#include <hip/hip_runtime.h>
#include <hip/hip_bf16.h>

// ---------------- constants ----------------
#define BN   8
#define CIN_ 16
#define HID_ 64
#define TT_  4
#define TAU_ 5
#define PSP  4096                       // T*W*H = 4*32*32
#define NSP  ((size_t)BN * HID_ * PSP)  // elements per [B,HID,T,W,H] tensor

// async-to-LDS availability (probe-guarded: falls back to VGPR bounce + ds_store)
#if defined(__has_builtin)
#  if __has_builtin(__builtin_amdgcn_global_load_async_to_lds_b128)
#    define HAVE_ASYNC 1
#  endif
#endif
#ifndef HAVE_ASYNC
#  define HAVE_ASYNC 0
#endif

#define AS1 __attribute__((address_space(1)))
#define AS3 __attribute__((address_space(3)))
typedef int v4i_vs __attribute__((vector_size(16)));   // matches builtin's V4i pointee

// ---------------- WMMA fragment types ----------------
typedef __attribute__((ext_vector_type(16))) __bf16 v16bf;
typedef __attribute__((ext_vector_type(8)))  float  v8f;

struct __align__(16) Q4 { unsigned x, y, z, w; };
union ABfrag { v16bf v; Q4 q[2]; };
union Cfrag  { v8f  v; Q4 q[2]; float f[8]; };

__device__ __forceinline__ void wait_async_lds()
{
#if HAVE_ASYNC
#  if __has_builtin(__builtin_amdgcn_s_wait_asynccnt)
    __builtin_amdgcn_s_wait_asynccnt(0);
#  else
    asm volatile("s_wait_asynccnt 0x0" ::: "memory");
#  endif
#endif
}

// ---- direct-global GEMM step: NCH chunks of K=32 bf16 from global input ----
// A layout (16-bit A 16x32): lane l -> row M=l&15; element j -> cin = ch*32 + (j<8 ? hi*8+j : 16+hi*8+j-8)
template<int NCH>
__device__ __forceinline__ void gemm_in(const __bf16* __restrict__ in, size_t si, int C,
                                        const __bf16* __restrict__ wp_tap,
                                        int nt, int lane, int hi, bool inb, Cfrag& acc)
{
    const __bf16* sp = in + si * (size_t)C;
#pragma unroll
    for (int ch = 0; ch < NCH; ++ch) {
        Q4 z = {0u, 0u, 0u, 0u};
        ABfrag a;
        const int c0 = ch * 32 + hi * 8;
        const int c1 = c0 + 16;
        a.q[0] = (inb && c0 < C) ? *(const Q4*)(sp + c0) : z;
        a.q[1] = (inb && c1 < C) ? *(const Q4*)(sp + c1) : z;
        ABfrag bf;
        const __bf16* wp = wp_tap + ((size_t)(ch * 4 + nt)) * 512 + lane * 16;
        bf.q[0] = *(const Q4*)wp;
        bf.q[1] = *(const Q4*)(wp + 8);
        acc.v = __builtin_amdgcn_wmma_f32_16x16x32_bf16(false, a.v, false, bf.v,
                                                        (short)0, acc.v, false, false);
    }
}

// ---- LDS-staged GEMM step: A fragments via ds_load_b128 from halo tile ----
template<int C, int NCH>
__device__ __forceinline__ void gemm_lds(const __bf16* tile, int row, int hpos,
                                         const __bf16* __restrict__ wp_tap,
                                         int nt, int lane, int hi, Cfrag& acc)
{
    const __bf16* sp = tile + (row * 34 + hpos) * C;
#pragma unroll
    for (int ch = 0; ch < NCH; ++ch) {
        Q4 z = {0u, 0u, 0u, 0u};
        ABfrag a;
        const int c0 = ch * 32 + hi * 8;
        const int c1 = c0 + 16;
        a.q[0] = (c0 < C) ? *(const Q4*)(sp + c0) : z;
        a.q[1] = (c1 < C) ? *(const Q4*)(sp + c1) : z;
        ABfrag bf;
        const __bf16* wp = wp_tap + ((size_t)(ch * 4 + nt)) * 512 + lane * 16;
        bf.q[0] = *(const Q4*)wp;
        bf.q[1] = *(const Q4*)(wp + 8);
        acc.v = __builtin_amdgcn_wmma_f32_16x16x32_bf16(false, a.v, false, bf.v,
                                                        (short)0, acc.v, false, false);
    }
}

// Fill 9x34xC halo tile for this block's h-row (async global->LDS; OOB chunks zeroed).
template<int C>
__device__ __forceinline__ void fill_tile(__bf16* tile, const __bf16* __restrict__ in,
                                          int b, int t0b, int w0b)
{
    constexpr int CB  = C / 8;       // 16-byte chunks per position
    constexpr int CPR = 34 * CB;     // chunks per halo row
    constexpr int TOT = 9 * CPR;
    for (int idx = threadIdx.x; idx < TOT; idx += 256) {
        const int row  = idx / CPR, rem = idx % CPR;
        const int hpos = rem / CB,  cblk = rem % CB;
        const int t = t0b + row / 3 - 1;
        const int w = w0b + row % 3 - 1;
        const int h = hpos - 1;
        const int loff = idx * 8;    // == (row*34+hpos)*C + cblk*8
        const bool valid = ((unsigned)t < (unsigned)TT_) && ((unsigned)w < 32u) && ((unsigned)h < 32u);
        if (valid) {
            const __bf16* g = in + ((size_t)b * PSP + (size_t)(t * 1024 + w * 32 + h)) * C + cblk * 8;
#if HAVE_ASYNC
            __builtin_amdgcn_global_load_async_to_lds_b128(
                (AS1 v4i_vs*)g, (AS3 v4i_vs*)(tile + loff), 0, 0);
#else
            *(Q4*)(tile + loff) = *(const Q4*)g;
#endif
        } else {
            Q4 z = {0u, 0u, 0u, 0u};
            *(Q4*)(tile + loff) = z;
        }
    }
}

// ---- 3x3x3 'SAME' implicit-GEMM conv, LDS-staged main input (CL channels),
// ---- optional fused direct-global input (CG channels, e.g. the x-branch).
// Block = 256 threads = 8 waves; tile = 32 positions (one h-row) x 64 Cout.
// Grid.x = B * (PSP/32) = 1024.
template<int NCHG, int NCHL, int CL>
__global__ __launch_bounds__(256) void conv_wmma_lds(
    const __bf16* __restrict__ inG, const __bf16* __restrict__ wpG, int CG,
    const __bf16* __restrict__ inL, const __bf16* __restrict__ wpL,
    const float* __restrict__ bias0, const float* __restrict__ bias1,
    float* __restrict__ out, int accum)
{
    __shared__ alignas(16) __bf16 tile[9 * 34 * CL];

    const int lane = threadIdx.x & 31;
    const int wv   = threadIdx.x >> 5;
    const int mt   = wv >> 2;           // 0..1 : M sub-tile
    const int nt   = wv & 3;            // 0..3 : N sub-tile
    const int b    = blockIdx.x >> 7;
    const int pt   = blockIdx.x & 127;
    const int pbase32 = pt * 32;
    const int t0b = pbase32 >> 10;
    const int w0b = (pbase32 >> 5) & 31;
    const int lrow = lane & 15, hi = lane >> 4;
    const int hloc = mt * 16 + lrow;    // h position of this lane's A row
    const int co   = nt * 16 + lrow;    // D layout: N = lane&15

    fill_tile<CL>(tile, inL, b, t0b, w0b);
    wait_async_lds();
    __syncthreads();

    Cfrag acc;
    float* op = out + (((size_t)b * HID_ + co) * PSP + pbase32 + mt * 16 + hi * 8);
    if (accum) { acc.q[0] = *(const Q4*)op; acc.q[1] = *(const Q4*)(op + 4); }
    else       { Q4 z = {0u,0u,0u,0u}; acc.q[0] = z; acc.q[1] = z; }

    for (int tap = 0; tap < 27; ++tap) {
        const int dt = tap / 9, dw = (tap / 3) % 3, dh = tap % 3;
        if (NCHG > 0) {
            const int tt = t0b + dt - 1, wy = w0b + dw - 1, hx = hloc + dh - 1;
            const bool inb = ((unsigned)tt < (unsigned)TT_) && ((unsigned)wy < 32u) && ((unsigned)hx < 32u);
            const size_t si = (size_t)b * PSP + (size_t)(tt * 1024 + wy * 32 + hx);
            gemm_in<(NCHG > 0 ? NCHG : 1)>(inG, si, CG, wpG + (size_t)tap * NCHG * 2048,
                                           nt, lane, hi, inb, acc);
        }
        gemm_lds<CL, NCHL>(tile, dt * 3 + dw, hloc + dh,
                           wpL + (size_t)tap * NCHL * 2048, nt, lane, hi, acc);
    }
    float bb = 0.f;
    if (bias0) bb += bias0[co];
    if (bias1) bb += bias1[co];
#pragma unroll
    for (int j = 0; j < 8; ++j) acc.f[j] += bb;   // rows M=j+8*hi -> contiguous p
    *(Q4*)op = acc.q[0];
    *(Q4*)(op + 4) = acc.q[1];
}

// ---- direct-global conv (used for the 1x1x1, which has no tap reuse) ----
template<int NTAPS, int NCHA, int NCHB>
__global__ __launch_bounds__(256) void conv_wmma(
    const __bf16* __restrict__ inA, const __bf16* __restrict__ wpA, int CA,
    const __bf16* __restrict__ inB, const __bf16* __restrict__ wpB, int CB,
    const float* __restrict__ bias0, const float* __restrict__ bias1,
    float* __restrict__ out, int accum)
{
    const int lane = threadIdx.x & 31;
    const int wv   = threadIdx.x >> 5;
    const int mt   = wv >> 2, nt = wv & 3;
    const int b    = blockIdx.x >> 7;
    const int pt   = blockIdx.x & 127;
    const int pbase = pt * 32 + mt * 16;
    const int lrow = lane & 15, hi = lane >> 4;
    const int p  = pbase + lrow;
    const int t0 = p >> 10, w0 = (p >> 5) & 31, h0 = p & 31;
    const int co = nt * 16 + lrow;

    Cfrag acc;
    float* op = out + (((size_t)b * HID_ + co) * PSP + pbase + hi * 8);
    if (accum) { acc.q[0] = *(const Q4*)op; acc.q[1] = *(const Q4*)(op + 4); }
    else       { Q4 z = {0u,0u,0u,0u}; acc.q[0] = z; acc.q[1] = z; }

    for (int tap = 0; tap < NTAPS; ++tap) {
        int tt = t0, wy = w0, hx = h0;
        if (NTAPS == 27) { tt += tap / 9 - 1; wy += (tap / 3) % 3 - 1; hx += tap % 3 - 1; }
        const bool inb = ((unsigned)tt < (unsigned)TT_) && ((unsigned)wy < 32u) && ((unsigned)hx < 32u);
        const size_t si = (size_t)b * PSP + (size_t)(tt * 1024 + wy * 32 + hx);
        gemm_in<NCHA>(inA, si, CA, wpA + (size_t)tap * NCHA * 2048, nt, lane, hi, inb, acc);
        if (NCHB > 0)
            gemm_in<NCHB>(inB, si, CB, wpB + (size_t)tap * NCHB * 2048, nt, lane, hi, inb, acc);
    }
    float bb = 0.f;
    if (bias0) bb += bias0[co];
    if (bias1) bb += bias1[co];
#pragma unroll
    for (int j = 0; j < 8; ++j) acc.f[j] += bb;
    *(Q4*)op = acc.q[0];
    *(Q4*)(op + 4) = acc.q[1];
}

// ---------------- prep kernels ----------------
// NCDHW f32 -> NDHWC bf16
__global__ void to_nhwc_k(const float* __restrict__ src, __bf16* __restrict__ dst,
                          int C, size_t total)
{
    size_t stride = (size_t)gridDim.x * blockDim.x;
    for (size_t i = (size_t)blockIdx.x * blockDim.x + threadIdx.x; i < total; i += stride) {
        int    c  = (int)(i % C);
        size_t bp = i / C;
        size_t pp = bp % PSP;
        size_t bb = bp / PSP;
        dst[i] = (__bf16)src[((size_t)bb * C + c) * PSP + pp];
    }
}

// Pack OIDHW f32 weights into per-(conv,tap,chunk,ntile) B-fragment blocks of 512 bf16.
__global__ void pack_w_k(const float* __restrict__ src, __bf16* __restrict__ dst,
                         int ntaps, int nch, int cinStride, int cinOff, long long total)
{
    long long stride = (long long)gridDim.x * blockDim.x;
    for (long long e = (long long)blockIdx.x * blockDim.x + threadIdx.x; e < total; e += stride) {
        int j     = (int)(e & 15);
        int lane  = (int)((e >> 4) & 31);
        int ntile = (int)((e >> 9) & 3);
        long long r = e >> 11;
        int ch  = (int)(r % nch);   r /= nch;
        int tap = (int)(r % ntaps); r /= ntaps;
        int cv  = (int)r;
        int co  = ntile * 16 + (lane & 15);
        int K   = (j < 8) ? 8 * (lane >> 4) + j : 16 + 8 * (lane >> 4) + (j - 8);
        int cin = ch * 32 + K;
        float v = 0.f;
        if (cinOff + cin < cinStride)
            v = src[((size_t)(cv * HID_ + co) * cinStride + cinOff + cin) * ntaps + tap];
        dst[e] = (__bf16)v;
    }
}

// ---------------- normalization / gates ----------------
__global__ void ln3_stats_k(const float* __restrict__ s, float* __restrict__ st)
{
    const float* d = s + (size_t)blockIdx.x * PSP;
    float sm = 0.f, sq = 0.f;
    for (int i = threadIdx.x; i < PSP; i += 256) { float v = d[i]; sm += v; sq += v * v; }
    __shared__ float A[256], Bq[256];
    A[threadIdx.x] = sm; Bq[threadIdx.x] = sq; __syncthreads();
    for (int o = 128; o; o >>= 1) {
        if (threadIdx.x < (unsigned)o) { A[threadIdx.x] += A[threadIdx.x + o]; Bq[threadIdx.x] += Bq[threadIdx.x + o]; }
        __syncthreads();
    }
    if (!threadIdx.x) {
        float mu  = A[0] * (1.f / PSP);
        float var = Bq[0] * (1.f / PSP) - mu * mu;
        st[2 * blockIdx.x] = mu;
        st[2 * blockIdx.x + 1] = rsqrtf(var + 1e-5f);
    }
}

// In-place: s0,s1,s3,s5 -> sigmoid(ln3), s2,s4 -> tanh(ln3)
__global__ void act6_k(float* __restrict__ s, const float* __restrict__ st, size_t n)
{
    size_t stride = (size_t)gridDim.x * blockDim.x;
    for (size_t i = (size_t)blockIdx.x * blockDim.x + threadIdx.x; i < n; i += stride) {
        size_t g = i >> 12;
        float  v = (s[i] - st[2 * g]) * st[2 * g + 1];
        size_t tn = i / NSP;
        s[i] = (tn == 2 || tn == 4) ? tanhf(v) : 1.f / (1.f + expf(-v));
    }
}

__global__ void scores_k(const float* __restrict__ r, const float* __restrict__ ch,
                         float* __restrict__ sc)
{
    int b = blockIdx.x / TAU_, l = blockIdx.x % TAU_;
    const float* rp = r + (size_t)b * HID_ * PSP;
    const float* cp = ch + ((size_t)l * BN + b) * HID_ * PSP;
    float s = 0.f;
    for (size_t i = threadIdx.x; i < (size_t)HID_ * PSP; i += 256) s += rp[i] * cp[i];
    __shared__ float A[256];
    A[threadIdx.x] = s; __syncthreads();
    for (int o = 128; o; o >>= 1) { if (threadIdx.x < (unsigned)o) A[threadIdx.x] += A[threadIdx.x + o]; __syncthreads(); }
    if (!threadIdx.x) sc[b * TAU_ + l] = A[0] * (1.f / 64.f);
}

// softmax over the BATCH axis (axis=0), per l — faithful to the reference
__global__ void softmax_k(const float* __restrict__ sc, float* __restrict__ at)
{
    if (threadIdx.x == 0 && blockIdx.x == 0) {
        for (int l = 0; l < TAU_; ++l) {
            float mx = -1e30f;
            for (int b = 0; b < BN; ++b) mx = fmaxf(mx, sc[b * TAU_ + l]);
            float e[BN], sm = 0.f;
            for (int b = 0; b < BN; ++b) { e[b] = expf(sc[b * TAU_ + l] - mx); sm += e[b]; }
            for (int b = 0; b < BN; ++b) at[b * TAU_ + l] = e[b] / sm;
        }
    }
}

__global__ void u_k(const float* __restrict__ ch, const float* __restrict__ at,
                    float* __restrict__ u, float* __restrict__ part)
{
    int g = blockIdx.x, b = g >> 6, c = g & 63;
    float a[TAU_];
    const float* base[TAU_];
    for (int l = 0; l < TAU_; ++l) {
        a[l] = at[b * TAU_ + l];
        base[l] = ch + (((size_t)l * BN + b) * HID_ + c) * PSP;
    }
    float* up = u + (size_t)g * PSP;
    float sm = 0.f, sq = 0.f;
    for (int p = threadIdx.x; p < PSP; p += 256) {
        float v = base[TAU_ - 1][p];
        for (int l = 0; l < TAU_; ++l) v += a[l] * base[l][p];
        up[p] = v; sm += v; sq += v * v;
    }
    __shared__ float A[256], Bq[256];
    A[threadIdx.x] = sm; Bq[threadIdx.x] = sq; __syncthreads();
    for (int o = 128; o; o >>= 1) {
        if (threadIdx.x < (unsigned)o) { A[threadIdx.x] += A[threadIdx.x + o]; Bq[threadIdx.x] += Bq[threadIdx.x + o]; }
        __syncthreads();
    }
    if (!threadIdx.x) { part[2 * g] = A[0]; part[2 * g + 1] = Bq[0]; }
}

__global__ void ln4_k(const float* __restrict__ part, float* __restrict__ st)
{
    int b = blockIdx.x;
    __shared__ float A[64], Bq[64];
    A[threadIdx.x]  = part[2 * (b * 64 + threadIdx.x)];
    Bq[threadIdx.x] = part[2 * (b * 64 + threadIdx.x) + 1];
    __syncthreads();
    for (int o = 32; o; o >>= 1) {
        if (threadIdx.x < (unsigned)o) { A[threadIdx.x] += A[threadIdx.x + o]; Bq[threadIdx.x] += Bq[threadIdx.x + o]; }
        __syncthreads();
    }
    if (!threadIdx.x) {
        const float inv = 1.f / ((float)HID_ * PSP);
        float mu  = A[0] * inv;
        float var = Bq[0] * inv - mu * mu;
        st[2 * b] = mu; st[2 * b + 1] = rsqrtf(var + 1e-5f);
    }
}

__global__ void cm_k(const float* __restrict__ s, const float* __restrict__ u,
                     const float* __restrict__ st4, const float* __restrict__ lnw,
                     const float* __restrict__ lnb, const float* __restrict__ m,
                     float* __restrict__ oc, float* __restrict__ om,
                     __bf16* __restrict__ cT, __bf16* __restrict__ mnT)
{
    size_t stride = (size_t)gridDim.x * blockDim.x;
    for (size_t i = (size_t)blockIdx.x * blockDim.x + threadIdx.x; i < NSP; i += stride) {
        int    p  = (int)(i & 4095);
        size_t bc = i >> 12;
        int    c  = (int)(bc & 63);
        int    b  = (int)(bc >> 6);
        float ln = (u[i] - st4[2 * b]) * st4[2 * b + 1] * lnw[c * PSP + p] + lnb[c * PSP + p];
        float cv = s[NSP + i] * s[2 * NSP + i] + ln;
        float mn = s[3 * NSP + i] * s[4 * NSP + i] + s[5 * NSP + i] * m[i];
        oc[i] = cv; om[i] = mn;
        size_t ti = ((size_t)b * PSP + p) * HID_ + c;
        cT[ti]  = (__bf16)cv;
        mnT[ti] = (__bf16)mn;
    }
}

__global__ void copy_k(const float* __restrict__ src, float* __restrict__ dst, size_t n)
{
    size_t stride = (size_t)gridDim.x * blockDim.x;
    for (size_t i = (size_t)blockIdx.x * blockDim.x + threadIdx.x; i < n; i += stride)
        dst[i] = src[i];
}

__global__ void hout_k(const float* __restrict__ s6, const float* __restrict__ st6,
                       const float* __restrict__ y, float* __restrict__ oh)
{
    size_t stride = (size_t)gridDim.x * blockDim.x;
    for (size_t i = (size_t)blockIdx.x * blockDim.x + threadIdx.x; i < NSP; i += stride) {
        size_t g = i >> 12;
        float  o = 1.f / (1.f + expf(-(s6[i] - st6[2 * g]) * st6[2 * g + 1]));
        oh[i] = o * tanhf(y[i]);
    }
}

// ---------------- host ----------------
extern "C" void kernel_launch(void* const* d_in, const int* in_sizes, int n_in,
                              void* d_out, int out_size, void* d_ws, size_t ws_size,
                              hipStream_t stream)
{
    const float* x    = (const float*)d_in[0];
    const float* ch   = (const float*)d_in[1];
    const float* m    = (const float*)d_in[2];
    const float* h    = (const float*)d_in[3];
    const float* Wx   = (const float*)d_in[4];
    const float* bx   = (const float*)d_in[5];
    const float* Wh   = (const float*)d_in[6];
    const float* bh   = (const float*)d_in[7];
    const float* w111 = (const float*)d_in[8];
    const float* b111 = (const float*)d_in[9];
    const float* lnw  = (const float*)d_in[10];
    const float* lnb  = (const float*)d_in[11];
    float* out = (float*)d_out;
    (void)in_sizes; (void)n_in; (void)out_size; (void)ws_size;

    char* ws = (char*)d_ws;
    size_t off = 0;
    auto alloc = [&](size_t bytes) -> char* {
        char* p = ws + off;
        off = (off + bytes + 255) & ~(size_t)255;
        return p;
    };
    float*  S    = (float*)alloc(7 * NSP * 4);
    float*  Y111 = (float*)alloc(NSP * 4);
    float*  U    = (float*)alloc(NSP * 4);
    __bf16* XT   = (__bf16*)alloc((size_t)BN * PSP * CIN_ * 2);
    __bf16* HT   = (__bf16*)alloc((size_t)BN * PSP * HID_ * 2);
    __bf16* MT   = (__bf16*)alloc((size_t)BN * PSP * HID_ * 2);
    __bf16* CT   = (__bf16*)alloc((size_t)BN * PSP * HID_ * 2);
    __bf16* MNT  = (__bf16*)alloc((size_t)BN * PSP * HID_ * 2);
    __bf16* WXP  = (__bf16*)alloc((size_t)7 * 27 * 1 * 2048 * 2);
    __bf16* WHP  = (__bf16*)alloc((size_t)9 * 27 * 2 * 2048 * 2);
    __bf16* W1C  = (__bf16*)alloc((size_t)2 * 2048 * 2);
    __bf16* W1M  = (__bf16*)alloc((size_t)2 * 2048 * 2);
    float*  ST3  = (float*)alloc(6 * 512 * 2 * 4);
    float*  ST6  = (float*)alloc(512 * 2 * 4);
    float*  PART = (float*)alloc(512 * 2 * 4);
    float*  ST4  = (float*)alloc(8 * 2 * 4);
    float*  SC   = (float*)alloc(BN * TAU_ * 4);
    float*  AT   = (float*)alloc(BN * TAU_ * 4);

    // ---- prep: NDHWC bf16 staging + weight fragment packing
    to_nhwc_k<<<2048, 256, 0, stream>>>(x, XT, CIN_, (size_t)BN * PSP * CIN_);
    to_nhwc_k<<<2048, 256, 0, stream>>>(h, HT, HID_, (size_t)BN * PSP * HID_);
    to_nhwc_k<<<2048, 256, 0, stream>>>(m, MT, HID_, (size_t)BN * PSP * HID_);
    pack_w_k<<<512,  256, 0, stream>>>(Wx,   WXP, 27, 1, CIN_,     0, 7LL * 27 * 2048);
    pack_w_k<<<1024, 256, 0, stream>>>(Wh,   WHP, 27, 2, HID_,     0, 9LL * 27 * 2 * 2048);
    pack_w_k<<<16,   256, 0, stream>>>(w111, W1C,  1, 2, 2 * HID_, 0,    2LL * 2048);
    pack_w_k<<<16,   256, 0, stream>>>(w111, W1M,  1, 2, 2 * HID_, HID_, 2LL * 2048);

    // ---- stage-1 convs: s_i = cx(i) + cs(state_i, i) + biases.
    // state operand LDS-staged (async halo tile), x-branch fused direct from L2.
    for (int i = 0; i < 7; ++i) {
        const __bf16* st = (i == 3 || i == 4 || i == 5) ? MT : HT;
        conv_wmma_lds<1, 2, HID_><<<1024, 256, 0, stream>>>(
            XT, WXP + (size_t)i * 27 * 2048, CIN_,
            st, WHP + (size_t)i * 27 * 2 * 2048,
            bx + i * HID_, bh + i * HID_, S + (size_t)i * NSP, 0);
    }

    // ---- layer norms + gates
    ln3_stats_k<<<6 * 512, 256, 0, stream>>>(S, ST3);
    act6_k<<<2048, 256, 0, stream>>>(S, ST3, 6 * NSP);

    // ---- attention recall + ln4 -> c, and m_new
    scores_k<<<BN * TAU_, 256, 0, stream>>>(S /* r = s0 */, ch, SC);
    softmax_k<<<1, 32, 0, stream>>>(SC, AT);
    u_k<<<512, 256, 0, stream>>>(ch, AT, U, PART);
    ln4_k<<<8, 64, 0, stream>>>(PART, ST4);
    cm_k<<<2048, 256, 0, stream>>>(S, U, ST4, lnw, lnb, m,
                                   out + 4 * NSP, out + 5 * NSP, CT, MNT);
    copy_k<<<4096, 256, 0, stream>>>(ch + NSP, out, 4 * NSP);  // c_history[1:]

    // ---- stage-2: s6 += cs(c,7) then += cs(m_new,8)  (LDS-staged, accumulate)
    conv_wmma_lds<0, 2, HID_><<<1024, 256, 0, stream>>>(
        nullptr, nullptr, 0,
        CT, WHP + (size_t)7 * 27 * 2 * 2048,
        bh + 7 * HID_, nullptr, S + 6 * NSP, 1);
    conv_wmma_lds<0, 2, HID_><<<1024, 256, 0, stream>>>(
        nullptr, nullptr, 0,
        MNT, WHP + (size_t)8 * 27 * 2 * 2048,
        bh + 8 * HID_, nullptr, S + 6 * NSP, 1);
    // y111 = conv1x1x1(concat(c, m_new)) + b111  (no tap reuse -> direct kernel)
    conv_wmma<1, 2, 2><<<1024, 256, 0, stream>>>(
        CT, W1C, HID_, MNT, W1M, HID_, b111, nullptr, Y111, 0);

    // ---- output gate + h_new
    ln3_stats_k<<<512, 256, 0, stream>>>(S + 6 * NSP, ST6);
    hout_k<<<2048, 256, 0, stream>>>(S + 6 * NSP, ST6, Y111, out + 6 * NSP);
}
